// RelationNetwork_11347303596677
// MI455X (gfx1250) — matrix operand order repaired and checked by reading, not compiled, not run on hardware
//
#include <hip/hip_runtime.h>
#include <hip/hip_bf16.h>
#include <stdint.h>

#define B_   32
#define N_   128
#define D_   64
#define REL_ 256
#define OUT_ 10
#define PAIRS_    (N_ * N_)        // 16384 per batch
#define ROWTILES_ (PAIRS_ / 16)    // 1024 per batch
#define SPLIT_    8                // row-tile splits per batch (atomic merge)

typedef __bf16 v16bf __attribute__((ext_vector_type(16)));
typedef __bf16 v8bf  __attribute__((ext_vector_type(8)));
typedef __bf16 v2bf  __attribute__((ext_vector_type(2)));
typedef float  v8f   __attribute__((ext_vector_type(8)));
typedef float  v2f   __attribute__((ext_vector_type(2)));

// ---------------------------------------------------------------------------
// Stage 0: zero the pooled accumulator (32 x 256 f32)
// ---------------------------------------------------------------------------
__global__ void rn_zero_kernel(float* __restrict__ p, int n) {
    int i = blockIdx.x * blockDim.x + threadIdx.x;
    if (i < n) p[i] = 0.f;
}

// ---------------------------------------------------------------------------
// Stage 1: P[b,n,:] = obj[b,n,:] @ Wg1[:64,:] + bg1 ;  Q[b,n,:] = obj @ Wg1[64:,:]
// One block per (b,n) row; 256 threads = one REL column each.
// ---------------------------------------------------------------------------
__global__ void rn_proj_kernel(const float* __restrict__ obj,
                               const float* __restrict__ Wg1,
                               const float* __restrict__ bg1,
                               float* __restrict__ Pb,
                               float* __restrict__ Q) {
    __shared__ float srow[D_];
    const int blk = blockIdx.x;          // b*N + n
    const int r   = threadIdx.x;         // 0..255
    if (r < D_) srow[r] = obj[blk * D_ + r];
    __syncthreads();
    float accP = bg1[r];
    float accQ = 0.f;
#pragma unroll 8
    for (int d = 0; d < D_; ++d) {
        float o = srow[d];
        accP += o * Wg1[d * REL_ + r];
        accQ += o * Wg1[(D_ + d) * REL_ + r];
    }
    Pb[(size_t)blk * REL_ + r] = accP;
    Q [(size_t)blk * REL_ + r] = accQ;
}

// ---------------------------------------------------------------------------
// Stage 2: fused  rel = ReLU( ReLU(P[i]+Q[j]) @ Wg2 + bg2 ), pooled += sum(rel)
// Block = 512 threads = 16 waves, one (batch, row-split) per block.
// Wave w owns output columns [16w, 16w+16); shared h-tile in LDS (bf16),
// double-buffered so only one barrier per tile is needed.
// ---------------------------------------------------------------------------
__global__ void __launch_bounds__(512)
rn_pair_gemm_kernel(const float* __restrict__ Pb,
                    const float* __restrict__ Q,
                    const float* __restrict__ Wg2,
                    const float* __restrict__ bg2,
                    float* __restrict__ pooled) {
    __shared__ __align__(16) uint16_t sh[2][16 * REL_];  // ping-pong h tiles (2x8 KB)

    const int b     = blockIdx.x / SPLIT_;
    const int split = blockIdx.x % SPLIT_;
    const int wave  = threadIdx.x >> 5;                // 0..15 -> column tile
    const int lane  = threadIdx.x & 31;
    const int half  = lane >> 4;                       // lane half (0/1)
    const int col   = wave * 16 + (lane & 15);         // output REL column

    // Preload this wave's Wg2 column slab as 8 bf16 B-fragments (64 VGPRs),
    // reused for all 128 row tiles.
    union Frag { v16bf v; uint32_t d[8]; int4 q[2]; };
    Frag bfrag[8];
#pragma unroll
    for (int s = 0; s < 8; ++s) {
#pragma unroll
        for (int e = 0; e < 16; e += 2) {              // B 32x16 bf16 layout
            int k = s * 32 + e + 16 * half;
            v2f t;
            t.x = Wg2[(size_t)k * REL_ + col];
            t.y = Wg2[(size_t)(k + 1) * REL_ + col];
            v2bf c = __builtin_convertvector(t, v2bf); // v_cvt_pk_bf16_f32
            bfrag[s].d[e >> 1] = __builtin_bit_cast(uint32_t, c);
        }
    }
    const float cbias = bg2[col];
    v8f runsum8 = {};                                  // vectorized pooled partials

    // h-tile build mapping: thread -> (row m, 8 consecutive K)
    const int hm = threadIdx.x >> 5;                   // 0..15
    const int hk = (threadIdx.x & 31) * 8;             // 0..248

    const float* Pbase = Pb + (size_t)(b * N_) * REL_;
    const float* Qbase = Q  + (size_t)(b * N_) * REL_;
    const int m = lane & 15;

    // This split owns pair rows i in [split*16, split*16+16), all 8 j-tiles each.
    for (int ic = 0; ic < N_ / SPLIT_; ++ic) {
        const int i = split * (N_ / SPLIT_) + ic;
        // P row chunk is invariant over the 8 inner j-tiles: hoist the loads.
        const float4* prow = (const float4*)(Pbase + (size_t)i * REL_);
        const float4 p0 = prow[hk / 4], p1 = prow[hk / 4 + 1];

        for (int jc = 0; jc < 8; ++jc) {
            const int j0 = jc * 16;
            const int pbuf = (ic * 8 + jc) & 1;        // ping-pong buffer
            uint16_t* hbuf = sh[pbuf];

            // Build h tile (writes race only with reads 2 tiles ago -> safe
            // across the single barrier below, thanks to double buffering).
            {
                const float4* qrow =
                    (const float4*)(Qbase + (size_t)(j0 + hm) * REL_);
                float4 q0 = qrow[hk / 4], q1 = qrow[hk / 4 + 1];
                if (jc + 1 < 8) {                      // prefetch next tile's Q row
                    __builtin_prefetch(
                        Qbase + (size_t)(j0 + 16 + hm) * REL_ + hk, 0, 1);
                }
                v8f v;
                v[0] = p0.x + q0.x; v[1] = p0.y + q0.y;
                v[2] = p0.z + q0.z; v[3] = p0.w + q0.w;
                v[4] = p1.x + q1.x; v[5] = p1.y + q1.y;
                v[6] = p1.z + q1.z; v[7] = p1.w + q1.w;
#pragma unroll
                for (int e = 0; e < 8; ++e)            // ReLU (g1)
                    v[e] = v[e] > 0.f ? v[e] : 0.f;
                v8bf hb = __builtin_convertvector(v, v8bf);  // packed f32->bf16 RNE
                *(int4*)&hbuf[hm * REL_ + hk] = __builtin_bit_cast(int4, hb);
            }
            __syncthreads();

            // 16x16 output tile: 8 chained bf16 WMMAs over K=256
            v8f acc = {};
#pragma unroll
            for (int s = 0; s < 8; ++s) {
                Frag af;
                const int k0 = s * 32;
                // A 16x32 bf16 layout: half selects K runs {0..15}/{16..31}
                af.q[0] = *(const int4*)&hbuf[m * REL_ + k0 + 16 * half];
                af.q[1] = *(const int4*)&hbuf[m * REL_ + k0 + 16 * half + 8];
                acc = __builtin_amdgcn_wmma_f32_16x16x32_bf16(
                    false, af.v, false, bfrag[s].v, (short)0, acc, false, false);
            }

            // bias + ReLU (g2); accumulate element-wise (packed, no serial chain)
#pragma unroll
            for (int vi = 0; vi < 8; ++vi) {
                float x = acc[vi] + cbias;
                runsum8[vi] += (x > 0.f ? x : 0.f);
            }
        }
    }

    // Reduce the 8 M-positions this lane holds, then combine lane halves
    // (M=0..7 with M=8..15) via ds_swizzle SWAPX16.
    float runsum = (((runsum8[0] + runsum8[1]) + (runsum8[2] + runsum8[3])) +
                    ((runsum8[4] + runsum8[5]) + (runsum8[6] + runsum8[7])));
    float other = __builtin_bit_cast(float,
        __builtin_amdgcn_ds_swizzle(__builtin_bit_cast(int, runsum), 0x401F));
    runsum += other;
    if (half == 0) {
        atomicAdd(&pooled[b * REL_ + col], runsum);    // merge SPLIT_ partials
    }
}

// ---------------------------------------------------------------------------
// Stage 3: out = ReLU(pooled @ Wf1 + bf1) @ Wf2 + bf2   (tiny)
// One block per batch, 256 threads.
// ---------------------------------------------------------------------------
__global__ void rn_head_kernel(const float* __restrict__ pooled,
                               const float* __restrict__ Wf1,
                               const float* __restrict__ bf1,
                               const float* __restrict__ Wf2,
                               const float* __restrict__ bf2,
                               float* __restrict__ out) {
    __shared__ float sp[REL_];
    __shared__ float shid[REL_];
    const int bb = blockIdx.x;
    const int t  = threadIdx.x;
    sp[t] = pooled[bb * REL_ + t];
    __syncthreads();
    float acc = bf1[t];
#pragma unroll 8
    for (int k = 0; k < REL_; ++k) acc += sp[k] * Wf1[k * REL_ + t];
    shid[t] = acc > 0.f ? acc : 0.f;
    __syncthreads();
    if (t < OUT_) {
        float o = bf2[t];
        for (int r = 0; r < REL_; ++r) o += shid[r] * Wf2[r * OUT_ + t];
        out[bb * OUT_ + t] = o;
    }
}

// ---------------------------------------------------------------------------
extern "C" void kernel_launch(void* const* d_in, const int* in_sizes, int n_in,
                              void* d_out, int out_size, void* d_ws, size_t ws_size,
                              hipStream_t stream) {
    const float* obj = (const float*)d_in[0];
    const float* Wg1 = (const float*)d_in[1];
    const float* bg1 = (const float*)d_in[2];
    const float* Wg2 = (const float*)d_in[3];
    const float* bg2 = (const float*)d_in[4];
    const float* Wf1 = (const float*)d_in[5];
    const float* bf1 = (const float*)d_in[6];
    const float* Wf2 = (const float*)d_in[7];
    const float* bf2 = (const float*)d_in[8];
    float* out = (float*)d_out;

    // Workspace layout: P (4 MB) | Q (4 MB) | pooled (32 KB)
    float* Pb     = (float*)d_ws;
    float* Q      = Pb + (size_t)B_ * N_ * REL_;
    float* pooled = Q  + (size_t)B_ * N_ * REL_;

    rn_zero_kernel<<<(B_ * REL_ + 255) / 256, 256, 0, stream>>>(pooled, B_ * REL_);
    rn_proj_kernel<<<B_ * N_, REL_, 0, stream>>>(obj, Wg1, bg1, Pb, Q);
    rn_pair_gemm_kernel<<<B_ * SPLIT_, 512, 0, stream>>>(Pb, Q, Wg2, bg2, pooled);
    rn_head_kernel<<<B_, REL_, 0, stream>>>(pooled, Wf1, bf1, Wf2, bf2, out);
}
